// RouteLearningModel_44306882625963
// MI455X (gfx1250) — matrix operand order repaired and checked by reading, not compiled, not run on hardware
//
#include <hip/hip_runtime.h>
#include <hip/hip_bf16.h>
#include <stdint.h>

// ---------------- model constants (match reference) ----------------
#define NGAT   200000      // N_PATH * SEQ
#define NPATH  10000
#define NOD    2000
#define ESEL   60000
#define HF     64          // H*F
#define NSLOPE 0.2f

typedef __attribute__((ext_vector_type(16))) __bf16 v16bf;
typedef __attribute__((ext_vector_type(8)))  __bf16 v8bf;
typedef __attribute__((ext_vector_type(8)))  float  v8f;

// ---- fp32 -> bf16 round-to-nearest-even
static __device__ __forceinline__ __bf16 f2bf(float f) {
  unsigned u = __builtin_bit_cast(unsigned, f);
  unsigned r = (u + 0x7FFFu + ((u >> 16) & 1u)) >> 16;
  return __builtin_bit_cast(__bf16, (unsigned short)r);
}
static __device__ __forceinline__ float bf2f(__bf16 b) {
  unsigned short u = __builtin_bit_cast(unsigned short, b);
  return __builtin_bit_cast(float, (unsigned)u << 16);
}

struct bfcat { v8bf lo, hi; };
static __device__ __forceinline__ v16bf cat16(v8bf lo, v8bf hi) {
  bfcat t{lo, hi};
  return __builtin_bit_cast(v16bf, t);
}

// ---- order-preserving f32 <-> u32 encoding for atomic max on floats
static __device__ __forceinline__ unsigned fenc(float f) {
  unsigned u = __builtin_bit_cast(unsigned, f);
  return (u >> 31) ? ~u : (u ^ 0x80000000u);
}
static __device__ __forceinline__ float fdec(unsigned e) {
  return (e >> 31) ? __builtin_bit_cast(float, e ^ 0x80000000u)
                   : __builtin_bit_cast(float, ~e);
}

// ====================================================================
// Weight pre-swizzle: f32 [K,N] -> bf16 in exact WMMA B-fragment order.
// Layout: [kt][nt][lane(32)][elem(16)], k = kt*32 + 8*(lane>>4) + (e&7) + 16*(e>>3),
// col = nt*16 + (lane&15). Zero-padded to Kpad x Npad.
// ====================================================================
__global__ void swizzle_w(const float* __restrict__ W, __bf16* __restrict__ out,
                          int K, int N, int Npad, int total)
{
  int tid = blockIdx.x * blockDim.x + threadIdx.x;
  if (tid >= total) return;
  int e    = tid & 15;
  int lane = (tid >> 4) & 31;
  int tile = tid >> 9;
  int ntiles = Npad >> 4;
  int nt = tile % ntiles, kt = tile / ntiles;
  int k   = kt * 32 + 8 * (lane >> 4) + (e & 7) + 16 * (e >> 3);
  int col = nt * 16 + (lane & 15);
  float v = (k < K && col < N) ? W[(size_t)k * N + col] : 0.0f;
  out[tid] = f2bf(v);
}

// plain f32 -> bf16 convert
__global__ void cvt_bf16(const float* __restrict__ in, __bf16* __restrict__ out, size_t n)
{
  size_t i = blockIdx.x * (size_t)blockDim.x + threadIdx.x;
  if (i < n) out[i] = f2bf(in[i]);
}

// hout += gat_bias (f32, in place) and emit bf16 copy for the MLP
__global__ void bias_cvt(float* __restrict__ hout, const float* __restrict__ bias,
                         __bf16* __restrict__ houtbf, size_t n)
{
  size_t i = blockIdx.x * (size_t)blockDim.x + threadIdx.x;
  if (i >= n) return;
  float v = hout[i] + bias[i & (HF - 1)];
  hout[i] = v;
  houtbf[i] = f2bf(v);
}

// ====================================================================
// WMMA GEMM, guard-free hot loop.
//   A:   bf16 [M, lda], lda % 32 == 0, zero-padded columns
//   Wsw: bf16 pre-swizzled fragments, Kpad x Npad, zero-padded
//   out: OutT [M, ldo] where ldo == Npad (all Npad cols written; pads = 0)
// One wave computes a 16x16 tile. Inner loop: 2x b128 A-loads,
// 1x 32B B-load, 1x v_wmma.
// ====================================================================
template <typename OutT>
__global__ void __launch_bounds__(32)
gemm_wmma(const __bf16* __restrict__ A, int lda,
          const __bf16* __restrict__ Wsw,
          const float* __restrict__ bias,       // [N] or nullptr
          OutT* __restrict__ out, int ldo,
          int N, int ktiles, int ntiles, int do_relu)
{
  const int lane = threadIdx.x & 31;
  const int half = lane >> 4;
  const int sub  = lane & 15;
  const int m0   = blockIdx.x * 16;
  const int n0   = blockIdx.y * 16;

  const __bf16* __restrict__ arow = A + (size_t)(m0 + sub) * lda + 8 * half;
  const __bf16* __restrict__ bptr = Wsw + ((size_t)blockIdx.y * 512) + lane * 16;
  const size_t bstride = (size_t)ntiles * 512;

  v8f acc = {0.f,0.f,0.f,0.f,0.f,0.f,0.f,0.f};
  for (int kt = 0; kt < ktiles; ++kt) {
    v8bf alo = *(const v8bf*)(arow);        // K = kt*32 + 8*half + 0..7
    v8bf ahi = *(const v8bf*)(arow + 16);   // K = kt*32 + 16 + 8*half + 0..7
    v16bf a = cat16(alo, ahi);
    v16bf b = *(const v16bf*)(bptr);
    acc = __builtin_amdgcn_wmma_f32_16x16x32_bf16(
        false, a, false, b, (short)0, acc, false, false);
    arow += 32;
    bptr += bstride;
  }

  const int col = n0 + sub;
  const float bv = (bias && col < N) ? bias[col] : 0.0f;
#pragma unroll
  for (int v = 0; v < 8; ++v) {
    int row = m0 + v + 8 * half;            // C/D layout: VGPR v -> row v + 8*half
    float r = acc[v] + bv;
    if (do_relu) r = r > 0.f ? r : 0.f;
    if constexpr (__is_same(OutT, float))
      out[(size_t)row * ldo + col] = r;
    else
      out[(size_t)row * ldo + col] = f2bf(r);
  }
}

// ====================================================================
// el/er projections: el[n,h] = sum_f x[n,h,f]*attn_l[h,f]
// ====================================================================
__global__ void attn_proj(const float* __restrict__ x,
                          const float* __restrict__ attn_l,
                          const float* __restrict__ attn_r,
                          float* __restrict__ el, float* __restrict__ er,
                          int ntot)
{
  int tid = blockIdx.x * blockDim.x + threadIdx.x;
  if (tid >= ntot) return;
  int n = tid >> 2, h = tid & 3;
  const float* xs = x + (size_t)n * HF + h * 16;
  float sl = 0.f, sr = 0.f;
#pragma unroll
  for (int f = 0; f < 16; ++f) {
    float v = xs[f];
    sl += v * attn_l[h * 16 + f];
    sr += v * attn_r[h * 16 + f];
  }
  el[tid] = sl;
  er[tid] = sr;
}

// ====================================================================
// Edge softmax passes
// ====================================================================
__global__ void edge_logit_max(const int* __restrict__ src, const int* __restrict__ dst,
                               const float* __restrict__ el, const float* __restrict__ er,
                               float* __restrict__ ebuf, unsigned* __restrict__ mbuf, int E)
{
  int i = blockIdx.x * blockDim.x + threadIdx.x;
  if (i >= E) return;
  int s = src[i], d = dst[i];
  const float4 lv = *(const float4*)(el + 4 * (size_t)s);
  const float4 rv = *(const float4*)(er + 4 * (size_t)d);
  float v[4] = { lv.x + rv.x, lv.y + rv.y, lv.z + rv.z, lv.w + rv.w };
#pragma unroll
  for (int h = 0; h < 4; ++h) {
    float e = v[h] > 0.f ? v[h] : NSLOPE * v[h];   // leaky_relu
    ebuf[4 * (size_t)i + h] = e;
    atomicMax(mbuf + 4 * (size_t)d + h, fenc(e));
  }
}

__global__ void edge_exp_sum(const int* __restrict__ dst,
                             float* __restrict__ ebuf,
                             const unsigned* __restrict__ mbuf,
                             float* __restrict__ sbuf, int E)
{
  int i = blockIdx.x * blockDim.x + threadIdx.x;
  if (i >= E) return;
  int d = dst[i];
#pragma unroll
  for (int h = 0; h < 4; ++h) {
    float m = fdec(mbuf[4 * (size_t)d + h]);
    float p = __expf(ebuf[4 * (size_t)i + h] - m);
    ebuf[4 * (size_t)i + h] = p;
    atomicAdd(sbuf + 4 * (size_t)d + h, p);
  }
}

// one thread per (edge, head): scatter x[src,h,:]*a into h_out[dst,h,:]
__global__ void edge_aggregate(const int* __restrict__ src, const int* __restrict__ dst,
                               const float* __restrict__ ebuf, const float* __restrict__ sbuf,
                               const float* __restrict__ x, float* __restrict__ hout, int E4)
{
  int t = blockIdx.x * blockDim.x + threadIdx.x;
  if (t >= E4) return;
  int i = t >> 2, h = t & 3;
  int s = src[i], d = dst[i];
  float w = ebuf[4 * (size_t)i + h] / sbuf[4 * (size_t)d + h];
  const float* xs = x + (size_t)s * HF + h * 16;
  float* ho = hout + (size_t)d * HF + h * 16;
#pragma unroll
  for (int f = 0; f < 16; ++f) atomicAdd(ho + f, w * xs[f]);
}

// ====================================================================
// Final MLP layer (32->1) fused with w_trans GEMV; one wave per path row
// ====================================================================
__global__ void final_emb(const __bf16* __restrict__ z6,    // [NPATH,32] bf16
                          const float* __restrict__ w6,     // [32]
                          const float* __restrict__ b6,     // [1]
                          const float* __restrict__ gat,    // [NPATH,1280] f32
                          const float* __restrict__ wtr,    // [1280]
                          float* __restrict__ gemb)
{
  int row  = blockIdx.x * (blockDim.x >> 5) + (threadIdx.x >> 5);
  int lane = threadIdx.x & 31;
  if (row >= NPATH) return;
  float acc = bf2f(z6[(size_t)row * 32 + lane]) * w6[lane];
  const float* g = gat + (size_t)row * 1280;
  for (int j = lane; j < 1280; j += 32) acc += g[j] * wtr[j];
#pragma unroll
  for (int o = 16; o > 0; o >>= 1) acc += __shfl_xor(acc, o, 32);
  if (lane == 0) gemb[row] = acc + b6[0];
}

// ====================================================================
// OD segment softmax + prediction
// ====================================================================
__global__ void od_logit_max(const int* __restrict__ sel_path, const int* __restrict__ sel_od,
                             const float* __restrict__ gemb,
                             float* __restrict__ lg, unsigned* __restrict__ mo, int n)
{
  int i = blockIdx.x * blockDim.x + threadIdx.x;
  if (i >= n) return;
  float v = gemb[sel_path[i]];
  lg[i] = v;
  atomicMax(mo + sel_od[i], fenc(v));
}

__global__ void od_exp_sum(const int* __restrict__ sel_od,
                           float* __restrict__ lg, const unsigned* __restrict__ mo,
                           float* __restrict__ so, int n)
{
  int i = blockIdx.x * blockDim.x + threadIdx.x;
  if (i >= n) return;
  int o = sel_od[i];
  float p = __expf(lg[i] - fdec(mo[o]));
  lg[i] = p;
  atomicAdd(so + o, p);
}

__global__ void od_final(const int* __restrict__ sel_path, const int* __restrict__ sel_od,
                         const float* __restrict__ lg, const float* __restrict__ so,
                         const float* __restrict__ odNum, float* __restrict__ out, int n)
{
  int i = blockIdx.x * blockDim.x + threadIdx.x;
  if (i >= n) return;
  int o = sel_od[i];
  float prob = lg[i] / so[o];
  out[NPATH + i] = prob;                              // prob output
  atomicAdd(out + sel_path[i], odNum[o] * prob);      // pred output
}

__global__ void fill_u32(unsigned* __restrict__ p, unsigned v, size_t n)
{
  size_t i = blockIdx.x * (size_t)blockDim.x + threadIdx.x;
  if (i < n) p[i] = v;
}

// ====================================================================
static inline int cdiv(long long a, long long b) { return (int)((a + b - 1) / b); }

extern "C" void kernel_launch(void* const* d_in, const int* in_sizes, int n_in,
                              void* d_out, int out_size, void* d_ws, size_t ws_size,
                              hipStream_t stream)
{
  // ---- inputs (jax pytree flatten order: dict keys sorted; mlp = [(w,b)...]) ----
  const float* feat     = (const float*)d_in[0];   // [NGAT,64]
  const float* odNum    = (const float*)d_in[1];   // [NOD]
  const float* Wp       = (const float*)d_in[2];   // [64,64]
  const float* attn_l   = (const float*)d_in[3];   // [4,16]
  const float* attn_r   = (const float*)d_in[4];   // [4,16]
  const float* gat_bias = (const float*)d_in[5];   // [64]
  const float* mw[7], *mb[7];
  for (int i = 0; i < 7; ++i) { mw[i] = (const float*)d_in[6 + 2*i]; mb[i] = (const float*)d_in[7 + 2*i]; }
  const float* w_trans  = (const float*)d_in[20];  // [1280]
  const int* gat_src    = (const int*)d_in[21];
  const int* gat_dst    = (const int*)d_in[22];
  const int* sel_path   = (const int*)d_in[23];
  const int* sel_od     = (const int*)d_in[24];
  const int  E          = in_sizes[21];
  float* out = (float*)d_out;

  // ---- MLP padded geometry: K/N rounded so the GEMM loop is guard-free ----
  // layer:      L0    L1    L2    L3    L4    L5
  const int Kd[6]   = {1280, 180, 150, 128,  80,  64};   // true K
  const int Nd[6]   = { 180, 150, 128,  80,  64,  32};   // true N
  const int Kpad[6] = {1280, 192, 160, 128,  96,  64};   // == prev Npad, %32==0
  const int Npad[6] = { 192, 160, 128,  96,  64,  32};   // %32==0 (ldo of layer out)

  // ---- workspace carving ----
  uint8_t* ws = (uint8_t*)d_ws;
  size_t off = 0;
  auto carve = [&](size_t bytes) -> void* {
    void* p = ws + off;
    off = (off + bytes + 255) & ~(size_t)255;
    return p;
  };
  float*    x      = (float*)   carve((size_t)NGAT * HF * 4);   // projected feats (f32)
  float*    hout   = (float*)   carve((size_t)NGAT * HF * 4);   // aggregated == gat view (f32)
  __bf16*   featbf = (__bf16*)  carve((size_t)NGAT * HF * 2);
  __bf16*   houtbf = (__bf16*)  carve((size_t)NGAT * HF * 2);
  float*    el     = (float*)   carve((size_t)NGAT * 4 * 4);
  float*    er     = (float*)   carve((size_t)NGAT * 4 * 4);
  unsigned* mbuf   = (unsigned*)carve((size_t)NGAT * 4 * 4);
  float*    sbuf   = (float*)   carve((size_t)NGAT * 4 * 4);
  float*    ebuf   = (float*)   carve((size_t)E * 4 * 4);
  __bf16*   zb0    = (__bf16*)  carve((size_t)NPATH * 192 * 2);
  __bf16*   zb1    = (__bf16*)  carve((size_t)NPATH * 192 * 2);
  float*    gemb   = (float*)   carve((size_t)NPATH * 4);
  float*    odlg   = (float*)   carve((size_t)ESEL * 4);
  unsigned* odm    = (unsigned*)carve((size_t)NOD * 4);
  float*    ods    = (float*)   carve((size_t)NOD * 4);
  __bf16*   wsw0   = (__bf16*)  carve((size_t)64 * 64 * 2);     // swizzled GAT W
  __bf16*   wswm[6];
  for (int i = 0; i < 6; ++i) wswm[i] = (__bf16*)carve((size_t)Kpad[i] * Npad[i] * 2);
  (void)ws_size; (void)n_in; (void)out_size;

  const int B = 256;

  // ---- init scratch + output accumulators ----
  fill_u32<<<cdiv((size_t)NGAT * HF, B), B, 0, stream>>>((unsigned*)hout, 0u, (size_t)NGAT * HF);
  fill_u32<<<cdiv((size_t)NGAT * 4, B), B, 0, stream>>>(mbuf, 0u, (size_t)NGAT * 4); // enc-min
  fill_u32<<<cdiv((size_t)NGAT * 4, B), B, 0, stream>>>((unsigned*)sbuf, 0u, (size_t)NGAT * 4);
  fill_u32<<<cdiv(NOD, B), B, 0, stream>>>(odm, 0u, NOD);
  fill_u32<<<cdiv(NOD, B), B, 0, stream>>>((unsigned*)ods, 0u, NOD);
  fill_u32<<<cdiv(NPATH, B), B, 0, stream>>>((unsigned*)out, 0u, NPATH);

  // ---- one-time operand prep (tiny) ----
  swizzle_w<<<cdiv(64 * 64, B), B, 0, stream>>>(Wp, wsw0, 64, 64, 64, 64 * 64);
  for (int i = 0; i < 6; ++i) {
    int tot = Kpad[i] * Npad[i];
    swizzle_w<<<cdiv(tot, B), B, 0, stream>>>(mw[i], wswm[i], Kd[i], Nd[i], Npad[i], tot);
  }
  cvt_bf16<<<cdiv((size_t)NGAT * HF, B), B, 0, stream>>>(feat, featbf, (size_t)NGAT * HF);

  // ---- GATConv projection: x = feat @ W  (WMMA, f32 out) ----
  gemm_wmma<float><<<dim3(NGAT / 16, 4), 32, 0, stream>>>(
      featbf, HF, wsw0, nullptr, x, HF, HF, /*ktiles=*/2, /*ntiles=*/4, 0);

  // ---- attention logits per node ----
  attn_proj<<<cdiv((size_t)NGAT * 4, B), B, 0, stream>>>(x, attn_l, attn_r, el, er, NGAT * 4);

  // ---- edge softmax (3 passes) + weighted aggregation ----
  edge_logit_max<<<cdiv(E, B), B, 0, stream>>>(gat_src, gat_dst, el, er, ebuf, mbuf, E);
  edge_exp_sum  <<<cdiv(E, B), B, 0, stream>>>(gat_dst, ebuf, mbuf, sbuf, E);
  edge_aggregate<<<cdiv((size_t)E * 4, B), B, 0, stream>>>(gat_src, gat_dst, ebuf, sbuf, x, hout, E * 4);
  bias_cvt<<<cdiv((size_t)NGAT * HF, B), B, 0, stream>>>(hout, gat_bias, houtbf, (size_t)NGAT * HF);

  // ---- MLP chain (WMMA bf16 in / bf16 out, ReLU, zero-padded) ----
  const __bf16* Ain = houtbf;
  __bf16* zout[6] = { zb0, zb1, zb0, zb1, zb0, zb1 };
  for (int i = 0; i < 6; ++i) {
    gemm_wmma<__bf16><<<dim3(NPATH / 16, Npad[i] / 16), 32, 0, stream>>>(
        Ain, Kpad[i], wswm[i], mb[i], zout[i], Npad[i],
        Nd[i], Kpad[i] / 32, Npad[i] / 16, 1);
    Ain = zout[i];
  }
  // L6 (32 -> 1) fused with w_trans GEMV: gat_emb = z6@w6+b6 + gat@w_trans
  final_emb<<<cdiv(NPATH, 8), 256, 0, stream>>>(zb1, mw[6], mb[6], hout, w_trans, gemb);

  // ---- od2Path: segment softmax per od + prediction scatter ----
  od_logit_max<<<cdiv(ESEL, B), B, 0, stream>>>(sel_path, sel_od, gemb, odlg, odm, ESEL);
  od_exp_sum  <<<cdiv(ESEL, B), B, 0, stream>>>(sel_od, odlg, odm, ods, ESEL);
  od_final    <<<cdiv(ESEL, B), B, 0, stream>>>(sel_path, sel_od, odlg, ods, odNum, out, ESEL);
}
